// BasicDeconvolutionBlock_24266565222403
// MI455X (gfx1250) — compile-verified, matrix operand order checked by reading, not run on hardware
//
#include <hip/hip_runtime.h>

typedef unsigned short ushort_t;
typedef __attribute__((ext_vector_type(16))) __bf16 v16bf;
typedef __attribute__((ext_vector_type(8)))  float  v8f;

#define NN   100000
#define KK   27
#define INC  256
#define OUTC 128
#define PAIRS_PER_BLK 256            // 8 waves * 32 rows
#define NBLK_PAIR ((NN + PAIRS_PER_BLK - 1) / PAIRS_PER_BLK)   // 391
#define WK_ELEMS  (INC * OUTC)       // 32768 bf16 per offset k

struct U32x8 { uint4 lo, hi; };      // 32 bytes -> one v16bf

__device__ __forceinline__ ushort_t f2bf(float f) {
    unsigned u = __builtin_bit_cast(unsigned, f);
    unsigned r = u + 0x7FFFu + ((u >> 16) & 1u);   // round-to-nearest-even
    return (ushort_t)(r >> 16);
}

// ---------------- prep kernels ----------------

__global__ void zero_f4(float4* p, long n) {
    for (long i = (long)blockIdx.x * blockDim.x + threadIdx.x; i < n;
         i += (long)gridDim.x * blockDim.x)
        p[i] = make_float4(0.f, 0.f, 0.f, 0.f);
}

// x (N x 256 f32) -> xb ((N+1) x 256 bf16), row N zeroed (sentinel gather row)
__global__ void convert_x(const float* __restrict__ x, ushort_t* __restrict__ xb) {
    const long total = (long)(NN + 1) * INC;
    for (long i = (long)blockIdx.x * blockDim.x + threadIdx.x; i < total;
         i += (long)gridDim.x * blockDim.x)
        xb[i] = (i < (long)NN * INC) ? f2bf(x[i]) : (ushort_t)0;
}

// Pack W (27 x 256 x 128 f32) into WMMA B-fragment layout (bf16):
// block (k,s,t) = 512 bf16; lane holds 16 contiguous bf16 = 32B:
//   col n = t*16 + (lane&15)
//   K     = 32*s + (lane>=16 ? 8 : 0) + (e<8 ? e : e+8)
__global__ void pack_w(const float* __restrict__ W, ushort_t* __restrict__ wp) {
    const long total = (long)KK * WK_ELEMS;
    for (long o = (long)blockIdx.x * blockDim.x + threadIdx.x; o < total;
         o += (long)gridDim.x * blockDim.x) {
        int e    = (int)(o & 15);
        int lane = (int)((o >> 4) & 31);
        int t    = (int)((o >> 9) & 7);
        int s    = (int)((o >> 12) & 7);
        int k    = (int)(o >> 15);
        int K    = s * 32 + ((lane >= 16) ? 8 : 0) + (e < 8 ? e : e + 8);
        int n    = t * 16 + (lane & 15);
        wp[o] = f2bf(W[((long)k * INC + K) * OUTC + n]);
    }
}

// ---------------- main gather-GEMM-scatter ----------------

__global__ __launch_bounds__(256)
void spconv_gemm(const ushort_t* __restrict__ xb,      // (N+1) x 256 bf16
                 const ushort_t* __restrict__ wp,      // 27 x 32768 packed bf16
                 const int* __restrict__ in_idx,       // 27 x N
                 const int* __restrict__ out_idx,      // 27 x N
                 float* __restrict__ out)              // N x 128 f32
{
    __shared__ __align__(16) ushort_t lds_w[WK_ELEMS];   // 64 KB of 320 KB WGP LDS

    const int k        = blockIdx.y;
    const int blk_base = blockIdx.x * PAIRS_PER_BLK;

    // padding is tail-only: if the first pair of this block is the sentinel,
    // every pair in the block is -> uniform early exit, skip the LDS fill.
    if (in_idx[k * NN + blk_base] == NN) return;

    // cooperative stage of W_k into LDS (4096 x b128)
    {
        const uint4* src = (const uint4*)(wp + (long)k * WK_ELEMS);
        uint4*       dst = (uint4*)lds_w;
        for (int i = threadIdx.x; i < WK_ELEMS / 8; i += 256) dst[i] = src[i];
    }
    __syncthreads();

    const int wave = threadIdx.x >> 5;
    const int lane = threadIdx.x & 31;
    const int tile = blk_base + wave * 32;               // 32 gathered rows per wave
    if (tile >= NN) return;                              // no barriers after this point
    if (in_idx[k * NN + tile] == NN) return;             // whole 32-row tile is tail padding

    const int m  = lane & 15;            // A-matrix row M owned by this lane
    const int hi = lane >> 4;            // lane half selects K sub-chunk
    const int p0 = tile + m;
    const int p1 = tile + 16 + m;
    const int ii0 = in_idx[k * NN + p0];                 // gather rows (NN -> zero row)
    const int ii1 = in_idx[k * NN + p1];
    const int jj0 = out_idx[k * NN + p0];                // scatter rows (NN -> drop)
    const int jj1 = out_idx[k * NN + p1];

    const ushort_t* xrow0 = xb + (long)ii0 * INC;
    const ushort_t* xrow1 = xb + (long)ii1 * INC;

    v8f c0[8], c1[8];
    const v8f vzero = {0.f, 0.f, 0.f, 0.f, 0.f, 0.f, 0.f, 0.f};
#pragma unroll
    for (int t = 0; t < 8; ++t) { c0[t] = vzero; c1[t] = vzero; }

#pragma unroll
    for (int s = 0; s < 8; ++s) {
        // two A fragments (16x32 bf16 each): lane loads K=[base..base+7],[base+16..base+23]
        const int base = s * 32 + hi * 8;
        U32x8 a0u, a1u;
        a0u.lo = *(const uint4*)(xrow0 + base);
        a0u.hi = *(const uint4*)(xrow0 + base + 16);
        a1u.lo = *(const uint4*)(xrow1 + base);
        a1u.hi = *(const uint4*)(xrow1 + base + 16);
        const v16bf a0 = __builtin_bit_cast(v16bf, a0u);
        const v16bf a1 = __builtin_bit_cast(v16bf, a1u);
#pragma unroll
        for (int t = 0; t < 8; ++t) {
            const uint4* bp = (const uint4*)&lds_w[((s * 8 + t) * 32 + lane) * 16];
            U32x8 bu; bu.lo = bp[0]; bu.hi = bp[1];
            const v16bf b = __builtin_bit_cast(v16bf, bu);
            // two independent WMMAs per B fragment: back-to-back XDL issue, no RAW
            c0[t] = __builtin_amdgcn_wmma_f32_16x16x32_bf16(
                false, a0, false, b, (short)0, c0[t], false, false);
            c1[t] = __builtin_amdgcn_wmma_f32_16x16x32_bf16(
                false, a1, false, b, (short)0, c1[t], false, false);
        }
    }

    // scatter-add: C layout -> VGPR v holds M=v (lanes 0-15) / M=v+8 (lanes 16-31)
#pragma unroll
    for (int v = 0; v < 8; ++v) {
        const int r   = v + hi * 8;
        const int jr0 = __shfl(jj0, r, 32);    // lane r owns out row for M=r
        const int jr1 = __shfl(jj1, r, 32);
        if (jr0 < NN) {
            float* orow = out + (long)jr0 * OUTC + m;
#pragma unroll
            for (int t = 0; t < 8; ++t)
                atomicAdd(orow + t * 16, c0[t][v]);
        }
        if (jr1 < NN) {
            float* orow = out + (long)jr1 * OUTC + m;
#pragma unroll
            for (int t = 0; t < 8; ++t)
                atomicAdd(orow + t * 16, c1[t][v]);
        }
    }
}

// ---------------- batchnorm + leaky ReLU ----------------

__global__ __launch_bounds__(256)
void bn_stats(const float* __restrict__ out, float* __restrict__ sums) {
    const int c    = threadIdx.x & 127;
    const int half = threadIdx.x >> 7;
    float s = 0.f, q = 0.f;
    for (long r = (long)blockIdx.x * 2 + half; r < NN; r += (long)gridDim.x * 2) {
        float v = out[r * OUTC + c];
        s += v; q += v * v;
    }
    __shared__ float ls[256], lq[256];
    ls[threadIdx.x] = s; lq[threadIdx.x] = q;
    __syncthreads();
    if (threadIdx.x < 128) {
        atomicAdd(&sums[threadIdx.x],       ls[threadIdx.x] + ls[threadIdx.x + 128]);
        atomicAdd(&sums[128 + threadIdx.x], lq[threadIdx.x] + lq[threadIdx.x + 128]);
    }
}

__global__ void bn_apply(float* __restrict__ out, const float* __restrict__ sums,
                         const float* __restrict__ gamma, const float* __restrict__ beta) {
    const long total = (long)NN * OUTC;
    for (long i = (long)blockIdx.x * blockDim.x + threadIdx.x; i < total;
         i += (long)gridDim.x * blockDim.x) {
        const int c   = (int)(i & 127);
        const float mean = sums[c] * (1.f / NN);
        const float var  = sums[128 + c] * (1.f / NN) - mean * mean;
        float v = (out[i] - mean) * rsqrtf(var + 1e-5f) * gamma[c] + beta[c];
        out[i] = (v >= 0.f) ? v : 0.01f * v;
    }
}

// ---------------- launch ----------------

extern "C" void kernel_launch(void* const* d_in, const int* in_sizes, int n_in,
                              void* d_out, int out_size, void* d_ws, size_t ws_size,
                              hipStream_t stream) {
    const float* x       = (const float*)d_in[0];
    const float* W       = (const float*)d_in[1];
    const float* gamma   = (const float*)d_in[2];
    const float* beta    = (const float*)d_in[3];
    const int*   in_idx  = (const int*)d_in[4];
    const int*   out_idx = (const int*)d_in[5];
    float* out = (float*)d_out;

    // workspace layout
    char* ws = (char*)d_ws;
    constexpr size_t XB_BYTES = (size_t)(NN + 1) * INC * 2;   // 51,200,512 (512-aligned)
    constexpr size_t WP_BYTES = (size_t)KK * WK_ELEMS * 2;    //  1,769,472
    ushort_t* xb   = (ushort_t*)ws;
    ushort_t* wp   = (ushort_t*)(ws + XB_BYTES);
    float*    sums = (float*)(ws + XB_BYTES + WP_BYTES);      // 256 floats

    // 1) zero output accumulator + stats
    zero_f4<<<2048, 256, 0, stream>>>((float4*)out,  (long)NN * OUTC / 4);
    zero_f4<<<1,    256, 0, stream>>>((float4*)sums, 64);

    // 2) precision staging
    convert_x<<<4096, 256, 0, stream>>>(x, xb);
    pack_w<<<3456, 256, 0, stream>>>(W, wp);

    // 3) gather-GEMM(bf16 WMMA)-scatter over 27 offsets
    dim3 grid(NBLK_PAIR, KK);
    spconv_gemm<<<grid, 256, 0, stream>>>(xb, wp, in_idx, out_idx, out);

    // 4) batchnorm stats + fused normalize/leaky-ReLU
    bn_stats<<<1024, 256, 0, stream>>>(out, sums);
    bn_apply<<<4096, 256, 0, stream>>>(out, sums, gamma, beta);
}
